// SPDNet_16621523436010
// MI455X (gfx1250) — matrix operand order repaired and checked by reading, not compiled, not run on hardware
//
#include <hip/hip_runtime.h>

#define TDIM 1000
#define CDIM 62

typedef __attribute__((ext_vector_type(16))) __bf16 v16bf;
typedef __attribute__((ext_vector_type(8)))  float  v8f;

union Frag { v16bf v; unsigned u[8]; };
union Acc  { v8f v; float f[8]; };

// truncation split: x ~= hi + lo (both bf16); pack two elements' hi16 via v_perm
__device__ __forceinline__ void split_pack(float x0, float x1, unsigned* hi, unsigned* lo){
    unsigned b0 = __float_as_uint(x0), b1 = __float_as_uint(x1);
    *hi = __builtin_amdgcn_perm(b1, b0, 0x07060302u);        // (b1.hi16<<16)|b0.hi16
    float r0 = x0 - __uint_as_float(b0 & 0xffff0000u);
    float r1 = x1 - __uint_as_float(b1 & 0xffff0000u);
    *lo = __builtin_amdgcn_perm(__float_as_uint(r1), __float_as_uint(r0), 0x07060302u);
}

// C = A*B for 14x14 matrices stored with stride 17 in LDS (196 worker threads)
__device__ __forceinline__ void mm14(const float* A, const float* B, float* C, int tid){
    if (tid < 196){
        int i = tid / 14, j = tid % 14;
        float s = 0.f;
        #pragma unroll
        for (int k = 0; k < 14; ++k) s += A[i*17 + k] * B[k*17 + j];
        C[i*17 + j] = s;
    }
}

__global__ __launch_bounds__(256)
void spdnet_fused(const float* __restrict__ x,
                  const float* __restrict__ w1,
                  const float* __restrict__ w2,
                  const float* __restrict__ w3,
                  const float* __restrict__ fcw,
                  const float* __restrict__ fcb,
                  float* __restrict__ out)
{
    __shared__ float sQt[16*65];                        // Q^T padded 16x64
    __shared__ float sW23[14*14];
    __shared__ __attribute__((aligned(16))) unsigned sY[8][16*34]; // per-wave Y: u32 = lo16<<16 | hi16
    __shared__ float sRed[8][256];
    __shared__ float mY[16*17], mZ[16*17], mE[16*17], mY2[16*17], mZ2[16*17];

    const int tid  = threadIdx.x;
    const int wave = tid >> 5;
    const int lane = tid & 31;
    const int b    = blockIdx.x;
    const float* xb = x + (size_t)b * (CDIM * TDIM);

    // ---- W23 = w2 @ w3 ----
    if (tid < 196){
        int i = tid / 14, j = tid % 14;
        float s = 0.f;
        #pragma unroll
        for (int k = 0; k < 14; ++k) s += w2[i*14 + k] * w3[k*14 + j];
        sW23[i*14 + j] = s;
    }
    __syncthreads();
    // ---- Qt[m][k] = (w1@W23)^T padded; ones-row hook Qt[14][62]=1 ----
    for (int e = tid; e < 16*64; e += 256){
        int m = e >> 6, k = e & 63;
        float v = 0.f;
        if (m < 14 && k < CDIM){
            float s = 0.f;
            #pragma unroll
            for (int r = 0; r < 14; ++r) s += w1[k*14 + r] * sW23[r*14 + m];
            v = s;
        } else if (m == 14 && k == CDIM) v = 1.f;
        sQt[m*65 + k] = v;
    }
    __syncthreads();

    // ---- persistent Q^T A-fragments (2 k-steps, hi/lo) ----
    Frag qh[2], ql[2];
    {
        int m = lane & 15;
        int koff = (lane & 16) ? 16 : 0;
        #pragma unroll
        for (int s = 0; s < 2; ++s)
            #pragma unroll
            for (int v = 0; v < 8; ++v){
                int kb = 2*v + koff + 32*s;
                split_pack(sQt[m*65 + kb], sQt[m*65 + kb + 1], &qh[s].u[v], &ql[s].u[v]);
            }
    }

    Acc gHH, gHL, gLH;                 // 3 independent Gram chains
    #pragma unroll
    for (int v = 0; v < 8; ++v){ gHH.f[v] = 0.f; gHL.f[v] = 0.f; gLH.f[v] = 0.f; }

    for (int it = 0; it < 4; ++it){
        int tbase = (it*8 + wave) * 32;
        // ---- build all B fragments (2 tiles x 2 ksteps); unconditional clamped loads ----
        Frag bh[2][2], bl[2][2];
        {
            int n  = lane & 15;
            int kh = (lane & 16) ? 16 : 0;
            #pragma unroll
            for (int tt = 0; tt < 2; ++tt){
                int t = tbase + 16*tt + n;
                int  toff = (t < TDIM) ? t : 0;      // clamp address, keep loads branchless
                float msk = (t < TDIM) ? 1.f : 0.f;  // zero out padded columns
                #pragma unroll
                for (int s = 0; s < 2; ++s)
                    #pragma unroll
                    for (int v = 0; v < 8; ++v){
                        int k = 2*v + kh + 32*s;     // compile-time after unroll
                        float x0, x1;
                        if (k < CDIM)        x0 = msk * __builtin_nontemporal_load(&xb[(size_t)k*TDIM + toff]);
                        else if (k == CDIM)  x0 = msk;           // ones-row (mean hook)
                        else                 x0 = 0.f;
                        if (k + 1 < CDIM)    x1 = msk * __builtin_nontemporal_load(&xb[(size_t)(k+1)*TDIM + toff]);
                        else if (k+1 == CDIM) x1 = msk;
                        else                 x1 = 0.f;
                        split_pack(x0, x1, &bh[tt][s].u[v], &bl[tt][s].u[v]);
                    }
            }
        }
        // ---- projection: 4 independent chains (2 tiles x 2 accs), interleaved ----
        Acc yA[2], yB[2];
        #pragma unroll
        for (int v = 0; v < 8; ++v){ yA[0].f[v]=0.f; yA[1].f[v]=0.f; yB[0].f[v]=0.f; yB[1].f[v]=0.f; }
        yA[0].v = __builtin_amdgcn_wmma_f32_16x16x32_bf16(false, qh[0].v, false, bh[0][0].v, (short)0, yA[0].v, false, false);
        yA[1].v = __builtin_amdgcn_wmma_f32_16x16x32_bf16(false, qh[0].v, false, bh[1][0].v, (short)0, yA[1].v, false, false);
        yB[0].v = __builtin_amdgcn_wmma_f32_16x16x32_bf16(false, qh[0].v, false, bl[0][0].v, (short)0, yB[0].v, false, false);
        yB[1].v = __builtin_amdgcn_wmma_f32_16x16x32_bf16(false, qh[0].v, false, bl[1][0].v, (short)0, yB[1].v, false, false);
        yA[0].v = __builtin_amdgcn_wmma_f32_16x16x32_bf16(false, ql[0].v, false, bh[0][0].v, (short)0, yA[0].v, false, false);
        yA[1].v = __builtin_amdgcn_wmma_f32_16x16x32_bf16(false, ql[0].v, false, bh[1][0].v, (short)0, yA[1].v, false, false);
        yB[0].v = __builtin_amdgcn_wmma_f32_16x16x32_bf16(false, qh[1].v, false, bh[0][1].v, (short)0, yB[0].v, false, false);
        yB[1].v = __builtin_amdgcn_wmma_f32_16x16x32_bf16(false, qh[1].v, false, bh[1][1].v, (short)0, yB[1].v, false, false);
        yA[0].v = __builtin_amdgcn_wmma_f32_16x16x32_bf16(false, qh[1].v, false, bl[0][1].v, (short)0, yA[0].v, false, false);
        yA[1].v = __builtin_amdgcn_wmma_f32_16x16x32_bf16(false, qh[1].v, false, bl[1][1].v, (short)0, yA[1].v, false, false);
        yB[0].v = __builtin_amdgcn_wmma_f32_16x16x32_bf16(false, ql[1].v, false, bh[0][1].v, (short)0, yB[0].v, false, false);
        yB[1].v = __builtin_amdgcn_wmma_f32_16x16x32_bf16(false, ql[1].v, false, bh[1][1].v, (short)0, yB[1].v, false, false);

        // ---- stage Y (hi|lo packed u32) into this wave's private LDS region ----
        {
            int mb = (lane & 16) ? 8 : 0;       // C/D layout: VGPR v -> M=v(+8), N=lane&15
            int n  = lane & 15;
            #pragma unroll
            for (int tt = 0; tt < 2; ++tt)
                #pragma unroll
                for (int v = 0; v < 8; ++v){
                    float yv = yA[tt].f[v] + yB[tt].f[v];
                    unsigned bbits = __float_as_uint(yv);
                    float r = yv - __uint_as_float(bbits & 0xffff0000u);
                    sY[wave][(v + mb)*34 + 16*tt + n] =
                        __builtin_amdgcn_perm(__float_as_uint(r), bbits, 0x07060302u);
                }
        }
        // ---- Gram over this wave's 32 t's (same-wave LDS is in-order; no barrier) ----
        {
            Frag ah, al, gbh, gbl;
            const unsigned* Yw = sY[wave];
            int m = lane & 15;
            int koffA = (lane & 16) ? 16 : 0;
            int koffB = (lane & 16) ? 16 : 0;
            #pragma unroll
            for (int v = 0; v < 8; ++v){
                int ka = 2*v + koffA;
                unsigned long long wA = *(const unsigned long long*)&Yw[m*34 + ka];
                unsigned a0 = (unsigned)wA, a1 = (unsigned)(wA >> 32);
                ah.u[v] = __builtin_amdgcn_perm(a1, a0, 0x05040100u);  // hi16 pair
                al.u[v] = __builtin_amdgcn_perm(a1, a0, 0x07060302u);  // lo16 pair
                int kb = 2*v + koffB;
                unsigned long long wB = *(const unsigned long long*)&Yw[m*34 + kb];
                unsigned c0 = (unsigned)wB, c1 = (unsigned)(wB >> 32);
                gbh.u[v] = __builtin_amdgcn_perm(c1, c0, 0x05040100u);
                gbl.u[v] = __builtin_amdgcn_perm(c1, c0, 0x07060302u);
            }
            gHH.v = __builtin_amdgcn_wmma_f32_16x16x32_bf16(false, ah.v, false, gbh.v, (short)0, gHH.v, false, false);
            gHL.v = __builtin_amdgcn_wmma_f32_16x16x32_bf16(false, ah.v, false, gbl.v, (short)0, gHL.v, false, false);
            gLH.v = __builtin_amdgcn_wmma_f32_16x16x32_bf16(false, al.v, false, gbh.v, (short)0, gLH.v, false, false);
        }
    }

    // ---- reduce 8 partial Gram tiles ----
    {
        int mb = (lane & 16) ? 8 : 0, n = lane & 15;
        #pragma unroll
        for (int v = 0; v < 8; ++v)
            sRed[wave][(v + mb)*16 + n] = gHH.f[v] + gHL.f[v] + gLH.f[v];
    }
    __syncthreads();
    {
        int m = tid >> 4, n = tid & 15;
        float g = 0.f;
        #pragma unroll
        for (int w = 0; w < 8; ++w) g += sRed[w][m*16 + n];
        mY[m*17 + n] = g;
    }
    __syncthreads();
    // S = G/T - mu mu^T  (row 14 of G holds T*mu)
    if (tid < 196){
        int i = tid / 14, j = tid % 14;
        const float invT = 1.f / (float)TDIM;
        float mi = mY[14*17 + i] * invT;
        float mj = mY[14*17 + j] * invT;
        mE[i*17 + j] = mY[i*17 + j] * invT - mi * mj;
    }
    __syncthreads();
    if (tid < 196){
        int i = tid / 14, j = tid % 14;
        mY[i*17 + j] = mE[i*17 + j];          // Y0 = S
        mZ[i*17 + j] = (i == j) ? 1.f : 0.f;  // Z0 = I
    }
    __syncthreads();

    // ---- logm via 2x Newton-Schulz sqrt + 7-term Mercator ----
    for (int sq = 0; sq < 2; ++sq){
        int iters = (sq == 0) ? 6 : 5;
        for (int itn = 0; itn < iters; ++itn){
            mm14(mZ, mY, mE, tid); __syncthreads();           // E = Z*Y
            if (tid < 196){ int i = tid/14, j = tid%14;
                mE[i*17 + j] = ((i == j) ? 3.f : 0.f) - mE[i*17 + j]; }
            __syncthreads();
            mm14(mY, mE, mY2, tid);                           // Y2 = Y*(3I-ZY)
            mm14(mE, mZ, mZ2, tid);                           // Z2 = (3I-ZY)*Z
            __syncthreads();
            if (tid < 196){ int i = tid/14, j = tid%14;
                mY[i*17 + j] = 0.5f * mY2[i*17 + j];
                mZ[i*17 + j] = 0.5f * mZ2[i*17 + j]; }
            __syncthreads();
        }
        if (sq == 0){
            if (tid < 196){ int i = tid/14, j = tid%14;
                mZ[i*17 + j] = (i == j) ? 1.f : 0.f; }
            __syncthreads();
        }
    }
    // W = S^(1/4) - I ; P = c7*I ; matrix Horner for log(1+W)
    if (tid < 196){ int i = tid/14, j = tid%14;
        mE[i*17 + j] = mY[i*17 + j] - ((i == j) ? 1.f : 0.f);
        mZ[i*17 + j] = (i == j) ? (1.f/7.f) : 0.f; }
    __syncthreads();
    const float cs[6] = { -1.f/6.f, 1.f/5.f, -1.f/4.f, 1.f/3.f, -1.f/2.f, 1.f };
    for (int k = 0; k < 6; ++k){
        mm14(mE, mZ, mZ2, tid); __syncthreads();
        if (tid < 196){ int i = tid/14, j = tid%14;
            mZ[i*17 + j] = mZ2[i*17 + j] + ((i == j) ? cs[k] : 0.f); }
        __syncthreads();
    }
    mm14(mE, mZ, mZ2, tid); __syncthreads();   // Z2 = log(S^(1/4)); L = 4*Z2

    // ---- fc + sigmoid ----
    if (tid < 2){
        float s = fcb[tid];
        for (int i = 0; i < 14; ++i)
            #pragma unroll
            for (int j = 0; j < 14; ++j)
                s += fcw[tid*196 + i*14 + j] * (4.f * mZ2[i*17 + j]);
        out[(size_t)b*2 + tid] = 1.f / (1.f + expf(-s));
    }
}

extern "C" void kernel_launch(void* const* d_in, const int* in_sizes, int n_in,
                              void* d_out, int out_size, void* d_ws, size_t ws_size,
                              hipStream_t stream) {
    const float* x   = (const float*)d_in[0];
    const float* w1  = (const float*)d_in[1];
    const float* w2  = (const float*)d_in[2];
    const float* w3  = (const float*)d_in[3];
    const float* fcw = (const float*)d_in[4];
    const float* fcb = (const float*)d_in[5];
    float* out = (float*)d_out;
    int B = in_sizes[0] / (CDIM * TDIM);
    spdnet_fused<<<B, 256, 0, stream>>>(x, w1, w2, w3, fcw, fcb, out);
}
